// Marcher_17317308138012
// MI455X (gfx1250) — matrix operand order repaired and checked by reading, not compiled, not run on hardware
//
#include <hip/hip_runtime.h>
#include <stdint.h>

typedef __attribute__((ext_vector_type(16))) _Float16 v16h;
typedef __attribute__((ext_vector_type(2)))  _Float16 h2v;
typedef __attribute__((ext_vector_type(8)))  float    v8f;

#define HIDDEN 128
#define STEPS 32
#define WAVES_PER_BLOCK 8
#define RAYS_PER_WAVE 16
#define RAYS_PER_BLOCK (WAVES_PER_BLOCK * RAYS_PER_WAVE)

struct alignas(16) W1Pack { h2v x, y, z, b; };   // (W1x, W1y, W1z, b1) for a pair of hidden units

// Sphere-trace a 3->128->128->1 ReLU MLP, 32 dependent steps, fully on-chip.
// One wave owns a 16-ray tile. Layer-2 (~97% of FLOPs) runs on
// v_wmma_f32_16x16x32_f16 with W2 pre-swizzled in LDS into B-fragment order.
// W2 is staged global->LDS with CDNA5 async-to-LDS DMA (ASYNCcnt).
__global__ __launch_bounds__(256)
void marcher_kernel(const float* __restrict__ pos,
                    const float* __restrict__ dirg,
                    const float* __restrict__ W1,
                    const float* __restrict__ b1,
                    const float* __restrict__ W2,
                    const float* __restrict__ b2,
                    const float* __restrict__ W3,
                    const float* __restrict__ b3,
                    float* __restrict__ out)
{
    // W2 as f16 B-fragments: frag = c*8+t (K-chunk c, N-tile t), per-lane 16 halfs contiguous.
    __shared__ alignas(32) _Float16 sB[32][32][16];                   // 32 KB
    __shared__ alignas(16) float    stage[4096];                      // 16 KB async staging (one K-chunk)
    __shared__ W1Pack               sW1h[64];                         // 1 KB packed layer-1 weights
    __shared__ alignas(16) float    sP[WAVES_PER_BLOCK][16][4];       // ray positions per wave
    __shared__ alignas(16) float    sPart[WAVES_PER_BLOCK][2][16][8]; // layer-3 partials

    const int tid  = threadIdx.x;
    const int wave = tid >> 5;
    const int lane = tid & 31;   // wave32
    const int hi   = lane >> 4;  // lane half (selects K subset / row half)
    const int n    = lane & 15;

    // ---- stage W2 with async global->LDS DMA, then swizzle-convert to f16 B-frags ----
    // B 32x16 f16 layout: lanes 0-15 hold K=0..15, lanes 16-31 hold K=16..31;
    // slot e (= 2*VGPR+half) holds K = e + 16*laneHalf.
    for (int ch = 0; ch < 4; ++ch) {                 // K-chunk c == ch (32 rows of W2)
        for (int v = tid; v < 1024; v += 256) {
            uint64_t g = (uint64_t)(uintptr_t)(W2 + (ch << 12) + (v << 2));
            uint32_t l = (uint32_t)(uintptr_t)(&stage[v << 2]);  // LDS addr = low 32 bits of flat
            asm volatile("global_load_async_to_lds_b128 %0, %1, off"
                         :: "v"(l), "v"(g) : "memory");
        }
        asm volatile("s_wait_asynccnt 0x0" ::: "memory");
        __syncthreads();
        for (int idx = tid; idx < 4096; idx += 256) {
            int kk = idx >> 7, j = idx & 127;        // row-in-chunk, column
            float v = stage[idx];
            int bh = kk >> 4, e = kk & 15;
            sB[(ch << 3) + (j >> 4)][(j & 15) + (bh << 4)][e] = (_Float16)v;
        }
        __syncthreads();
    }

    // Layer-1 weights packed as f16 pairs (A-frag slots use consecutive-K pairs).
    for (int p = tid; p < 64; p += 256) {
        W1Pack wp;
        wp.x = h2v{(_Float16)W1[2 * p],       (_Float16)W1[2 * p + 1]};
        wp.y = h2v{(_Float16)W1[128 + 2 * p], (_Float16)W1[128 + 2 * p + 1]};
        wp.z = h2v{(_Float16)W1[256 + 2 * p], (_Float16)W1[256 + 2 * p + 1]};
        wp.b = h2v{(_Float16)b1[2 * p],       (_Float16)b1[2 * p + 1]};
        sW1h[p] = wp;
    }

    const int rayBase = blockIdx.x * RAYS_PER_BLOCK + wave * RAYS_PER_WAVE;
    float dx = 0.f, dy = 0.f, dz = 0.f;
    if (lane < 16) {
        int r = rayBase + lane;
        sP[wave][lane][0] = pos[3 * r + 0];
        sP[wave][lane][1] = pos[3 * r + 1];
        sP[wave][lane][2] = pos[3 * r + 2];
        sP[wave][lane][3] = 0.f;
        dx = dirg[3 * r + 0]; dy = dirg[3 * r + 1]; dz = dirg[3 * r + 2];
    }
    // Per-lane constants: W3 / b2 depend only on column n = lane&15.
    float w3v[8], b2v[8];
    #pragma unroll
    for (int t = 0; t < 8; ++t) { w3v[t] = W3[(t << 4) + n]; b2v[t] = b2[(t << 4) + n]; }
    const float b3s = b3[0];

    __syncthreads();

    for (int s = 0; s < STEPS; ++s) {
        // Every lane reads its row's position (lanes L and L+16 share row n).
        const float4 pr = *(const float4*)&sP[wave][n][0];
        const _Float16 phx = (_Float16)pr.x, phy = (_Float16)pr.y, phz = (_Float16)pr.z;
        const h2v px2 = h2v{phx, phx}, py2 = h2v{phy, phy}, pz2 = h2v{phz, phz};
        const h2v zero2 = h2v{(_Float16)0.f, (_Float16)0.f};

        // ---- layer 1: h1 = relu(p@W1 + b1), built directly in A-frag layout ----
        // A 16x32 f16: lane row M = n; slot e holds K = e + 16*hi.
        // Packed: slots (e, e+1), e even, are consecutive K -> one v_pk_fma chain per pair.
        v16h a[4];
        #pragma unroll
        for (int c = 0; c < 4; ++c) {
            #pragma unroll
            for (int ep = 0; ep < 8; ++ep) {
                int e  = ep << 1;
                int j0 = (c << 5) + e + (hi << 4);
                const W1Pack wp = sW1h[j0 >> 1];
                h2v r = __builtin_elementwise_fma(px2, wp.x,
                        __builtin_elementwise_fma(py2, wp.y,
                        __builtin_elementwise_fma(pz2, wp.z, wp.b)));
                r = __builtin_elementwise_max(r, zero2);
                a[c][e]     = r.x;
                a[c][e + 1] = r.y;
            }
        }

        // ---- layer 2 (WMMA) fused with layer-3 per-lane partial dot products ----
        float part[8];
        #pragma unroll
        for (int r = 0; r < 8; ++r) part[r] = 0.f;

        #pragma unroll
        for (int t = 0; t < 8; ++t) {
            v8f cc;
            #pragma unroll
            for (int r = 0; r < 8; ++r) cc[r] = b2v[t];   // bias pre-broadcast into C
            #pragma unroll
            for (int c = 0; c < 4; ++c) {
                const v16h bf = *(const v16h*)&sB[(c << 3) + t][lane][0];
                cc = __builtin_amdgcn_wmma_f32_16x16x32_f16(
                        false, a[c], false, bf, (short)0, cc, false, false);
            }
            // D layout: this lane holds rows M = r + 8*hi at column j = 16*t + n.
            #pragma unroll
            for (int r = 0; r < 8; ++r)
                part[r] = fmaf(fmaxf(cc[r], 0.f), w3v[t], part[r]);
        }

        // ---- layer-3 reduction across the 16 columns (per-wave LDS; DS is in-order) ----
        *(float4*)&sPart[wave][hi][n][0] = make_float4(part[0], part[1], part[2], part[3]);
        *(float4*)&sPart[wave][hi][n][4] = make_float4(part[4], part[5], part[6], part[7]);
        asm volatile("" ::: "memory");   // keep DS ops ordered at the compiler level
        if (lane < 16) {
            float ssum = 0.f;
            #pragma unroll
            for (int q = 0; q < 16; ++q)
                ssum += sPart[wave][lane >> 3][q][lane & 7];
            float d = ssum + b3s;        // sdf value for row = lane
            sP[wave][lane][0] = fmaf(d, dx, pr.x);
            sP[wave][lane][1] = fmaf(d, dy, pr.y);
            sP[wave][lane][2] = fmaf(d, dz, pr.z);
        }
        asm volatile("" ::: "memory");
    }

    if (lane < 16) {
        int r = rayBase + lane;
        out[3 * r + 0] = sP[wave][lane][0];
        out[3 * r + 1] = sP[wave][lane][1];
        out[3 * r + 2] = sP[wave][lane][2];
    }
}

extern "C" void kernel_launch(void* const* d_in, const int* in_sizes, int n_in,
                              void* d_out, int out_size, void* d_ws, size_t ws_size,
                              hipStream_t stream) {
    (void)n_in; (void)out_size; (void)d_ws; (void)ws_size;
    const float* pos = (const float*)d_in[0];
    const float* dir = (const float*)d_in[1];
    const float* W1  = (const float*)d_in[2];
    const float* b1  = (const float*)d_in[3];
    const float* W2  = (const float*)d_in[4];
    const float* b2  = (const float*)d_in[5];
    const float* W3  = (const float*)d_in[6];
    const float* b3  = (const float*)d_in[7];
    float* out = (float*)d_out;

    int nRays  = in_sizes[0] / 3;               // 65536
    int blocks = nRays / RAYS_PER_BLOCK;        // 512
    marcher_kernel<<<blocks, 256, 0, stream>>>(pos, dir, W1, b1, W2, b2, W3, b3, out);
}
